// VTS_88184268521772
// MI455X (gfx1250) — compile-verified, matrix operand order checked by reading, not run on hardware
//
#include <hip/hip_runtime.h>
#include <math.h>

// Problem dims (fixed by the reference)
#define BB   8
#define LL   1024
#define DD   2048
#define HH   512
#define OM   5
#define TT   (LL - 2*OM)      // 1014 center positions
#define MTOT (BB*TT)          // 8112 rows
#define KDB  (OM*DD)          // 10240
#define KR   (2*OM*DD)        // 20480
#define KJ   513
#define KJP  544              // 513 padded to multiple of 32
#define G4H  (4*HH)           // 2048 gate rows

typedef __attribute__((ext_vector_type(16))) __bf16 v16bf;
typedef __attribute__((ext_vector_type(8)))  __bf16 bf16x8;
typedef __attribute__((ext_vector_type(8)))  float  v8f;

#define CAT16(lo, hi) __builtin_shufflevector((lo),(hi),0,1,2,3,4,5,6,7,8,9,10,11,12,13,14,15)

// LDS tile row pitch: 32 K-elems padded to 40 (80B) so that 16 consecutive
// rows map to disjoint 4-bank groups (20*r mod 64 tiles all 64 banks).
#define TP 40

// ---------------------------------------------------------------------------
// Packing / conversion kernels
// ---------------------------------------------------------------------------
__global__ void k_cvt_bf16(const float* __restrict__ src, __bf16* __restrict__ dst, int n) {
  int stride = gridDim.x * blockDim.x;
  for (int i = blockIdx.x*blockDim.x + threadIdx.x; i < n; i += stride)
    dst[i] = (__bf16)src[i];
}

// src [H][D][taps]  ->  dst [H][taps*D] with k = j*D + d (matches contiguous x windows)
__global__ void k_pack_convw(const float* __restrict__ src, __bf16* __restrict__ dst, int taps) {
  int n = HH*DD*taps;
  int stride = gridDim.x * blockDim.x;
  for (int i = blockIdx.x*blockDim.x + threadIdx.x; i < n; i += stride) {
    int h = i / (DD*taps); int rem = i % (DD*taps);
    int d = rem / taps;    int j = rem % taps;
    dst[(size_t)h*taps*DD + (size_t)j*DD + d] = (__bf16)src[i];
  }
}

// src [4H][513] -> dst [4H][544] (zero padded K)
__global__ void k_pack_wih(const float* __restrict__ src, __bf16* __restrict__ dst) {
  int n = G4H*KJP;
  int stride = gridDim.x * blockDim.x;
  for (int i = blockIdx.x*blockDim.x + threadIdx.x; i < n; i += stride) {
    int r = i / KJP; int k = i % KJP;
    dst[i] = (k < KJ) ? (__bf16)src[(size_t)r*KJ + k] : (__bf16)0.f;
  }
}

__global__ void k_bias2(const float* __restrict__ a, const float* __restrict__ b,
                        float* __restrict__ o, int n) {
  int i = blockIdx.x*blockDim.x + threadIdx.x;
  if (i < n) o[i] = a[i] + b[i];
}

// zero recurrence counters + padded h-state every call (graph-replay safe)
__global__ void k_init(unsigned* __restrict__ cnt, __bf16* __restrict__ hcur) {
  int i = blockIdx.x*blockDim.x + threadIdx.x;
  if (i < 2048) cnt[i] = 0u;
  if (i < 2*16*HH) hcur[i] = (__bf16)0.f;
}

// ---------------------------------------------------------------------------
// LDS-staged bf16 WMMA GEMM:  Out[m][n] = sum_k A(m)[k] * Wp[n][k]  (+ bias[n])
// WG tile 64x256 (8 waves, wave tile 32x64), K stepped by 32 with a
// double-buffered LDS stage (A 64x32, B 256x32; pitch 40 = conflict-free).
// Global traffic: 20KB per K-step for 64 WMMAs (~0.31KB/WMMA) vs 48KB of
// per-wave redundant loads without staging. Software pipelined: global loads
// for step k+1 issue before the WMMAs of step k; LDS stores after; 1 barrier.
// Row m decomposes as (m/rowsPerB, m%rowsPerB) so the same kernel reads conv
// windows of x (contiguous K across taps*D) and plain row-major matrices.
// ---------------------------------------------------------------------------
__global__ __launch_bounds__(256) void k_gemm(
    const __bf16* __restrict__ A, long strideB, long strideP, int rowsPerB,
    int Mtotal, const __bf16* __restrict__ Wp, const float* __restrict__ bias,
    float* __restrict__ Out, int N, int K)
{
  __shared__ __bf16 At[2][64*TP];     //  2 x  5120 B
  __shared__ __bf16 Bt[2][256*TP];    //  2 x 20480 B  (51200 B total)

  const int tid  = threadIdx.x;
  const int lane = tid & 31;
  const int wid  = tid >> 5;
  const int wm = wid & 1, wn = wid >> 1;
  const int m0 = blockIdx.x*64;
  const int n0 = blockIdx.y*256;
  const int lr = lane & 15;
  const int kb = (lane >> 4) << 3;      // k-base per half-wave (A/B frag layout)

  // ---- loader assignments (cooperative, one b128 chunk = 8 bf16) ----
  const int ar  = tid >> 2;             // A row 0..63
  const int ak8 = (tid & 3) << 3;       // A k-chunk offset
  int arow = m0 + ar; if (arow >= Mtotal) arow = Mtotal - 1;   // clamp; never stored
  const __bf16* Abase =
      A + (long)(arow / rowsPerB)*strideB + (long)(arow % rowsPerB)*strideP + ak8;

  const __bf16* Bbase[4];
  int bnl[4], bk8[4];
#pragma unroll
  for (int i = 0; i < 4; ++i) {
    int c = tid + i*256;                // 1024 chunks: 256 rows x 4 chunks
    bnl[i] = c >> 2;
    bk8[i] = (c & 3) << 3;
    Bbase[i] = Wp + (size_t)(n0 + bnl[i]) * (size_t)K + bk8[i];
  }

  const v8f vzero = {0.f,0.f,0.f,0.f,0.f,0.f,0.f,0.f};
  v8f acc[2][4];
#pragma unroll
  for (int a = 0; a < 2; ++a)
#pragma unroll
    for (int b = 0; b < 4; ++b) acc[a][b] = vzero;

  const int KT = K >> 5;                // K / 32 (all K are multiples of 32)

  // prologue: stage tile 0
  *(bf16x8*)&At[0][ar*TP + ak8] = *(const bf16x8*)(Abase);
#pragma unroll
  for (int i = 0; i < 4; ++i)
    *(bf16x8*)&Bt[0][bnl[i]*TP + bk8[i]] = *(const bf16x8*)(Bbase[i]);
  __syncthreads();

  for (int kt = 0; kt < KT; ++kt) {
    const int cur = kt & 1, nxt = cur ^ 1;
    const bool more = (kt + 1) < KT;

    // issue next tile's global loads first (hide latency behind WMMAs)
    bf16x8 sa; bf16x8 sb[4];
    if (more) {
      const int ko = (kt + 1) << 5;
      sa = *(const bf16x8*)(Abase + ko);
#pragma unroll
      for (int i = 0; i < 4; ++i) sb[i] = *(const bf16x8*)(Bbase[i] + ko);
      if ((kt & 7) == 0 && (kt + 9) < KT) {
        __builtin_prefetch(Abase    + ((kt + 9) << 5), 0, 1);
        __builtin_prefetch(Bbase[0] + ((kt + 9) << 5), 0, 1);
      }
    }

    // compute from current LDS buffers
    v16bf av[2], bv[4];
#pragma unroll
    for (int a = 0; a < 2; ++a) {
      const __bf16* ap = &At[cur][(wm*32 + a*16 + lr)*TP + kb];
      bf16x8 lo = *(const bf16x8*)(ap);
      bf16x8 hi = *(const bf16x8*)(ap + 16);
      av[a] = CAT16(lo, hi);
    }
#pragma unroll
    for (int b = 0; b < 4; ++b) {
      const __bf16* bp = &Bt[cur][(wn*64 + b*16 + lr)*TP + kb];
      bf16x8 lo = *(const bf16x8*)(bp);
      bf16x8 hi = *(const bf16x8*)(bp + 16);
      bv[b] = CAT16(lo, hi);
    }
#pragma unroll
    for (int a = 0; a < 2; ++a)
#pragma unroll
      for (int b = 0; b < 4; ++b)
        acc[a][b] = __builtin_amdgcn_wmma_f32_16x16x32_bf16(
            false, av[a], false, bv[b], (short)0, acc[a][b], false, false);

    // stash next tile into the other buffer, then one barrier per K-step
    if (more) {
      *(bf16x8*)&At[nxt][ar*TP + ak8] = sa;
#pragma unroll
      for (int i = 0; i < 4; ++i)
        *(bf16x8*)&Bt[nxt][bnl[i]*TP + bk8[i]] = sb[i];
    }
    __syncthreads();
  }

  // epilogue
  const int mrow = 8*(lane >> 4);
#pragma unroll
  for (int b = 0; b < 4; ++b) {
    int n = n0 + wn*64 + b*16 + lr;
    float bia = bias ? bias[n] : 0.f;
#pragma unroll
    for (int a = 0; a < 2; ++a) {
#pragma unroll
      for (int i = 0; i < 8; ++i) {
        int m = m0 + wm*32 + a*16 + mrow + i;
        if (m < Mtotal) Out[(size_t)m*N + n] = acc[a][b][i] + bia;
      }
    }
  }
}

// ---------------------------------------------------------------------------
// vts_d dot + joint assembly (bf16, K padded to 544)
// ---------------------------------------------------------------------------
__global__ __launch_bounds__(256) void k_assemble(
    const float* __restrict__ db, const float* __restrict__ da,
    const float* __restrict__ r, __bf16* __restrict__ joint)
{
  __shared__ float red[256];
  int m = blockIdx.x, tid = threadIdx.x;
  float s = 0.f;
  for (int h = tid; h < HH; h += 256) s += db[(size_t)m*HH + h] * da[(size_t)m*HH + h];
  red[tid] = s; __syncthreads();
  for (int o = 128; o > 0; o >>= 1) { if (tid < o) red[tid] += red[tid+o]; __syncthreads(); }
  float vts = red[0];
  __bf16* jr = joint + (size_t)m*KJP;
  for (int h = tid; h < HH; h += 256) jr[h] = (__bf16)r[(size_t)m*HH + h];
  for (int k = HH + tid; k < KJP; k += 256) jr[k] = (__bf16)((k == HH) ? vts : 0.f);
}

// ---------------------------------------------------------------------------
// Persistent BiLSTM recurrence. 32 WGs: dir = blk/16, each WG owns 32 hidden
// units -> 128 Whh gate-rows pinned in LDS as bf16 (128KB; CDNA5 has 320KB).
// Per step: [16x512]x[512x128] bf16 WMMA (A = padded h state, B from LDS),
// f32 pointwise gates, then device-scope split barrier over 16 WGPs.
// ---------------------------------------------------------------------------
__global__ __launch_bounds__(256) void k_lstm(
    const float* __restrict__ Gx,       // [2][MTOT][4H]
    const __bf16* __restrict__ whh_pk,  // [2][4H][H]
    __bf16* __restrict__ hcur,          // [2][16][H] rows 8..15 stay zero
    float* __restrict__ hist,           // [2][B][T][H]
    unsigned* __restrict__ cnt)         // [2][1024] per-step arrival counters
{
  extern __shared__ char smem[];
  __bf16* whhS = (__bf16*)smem;                       // 128 x 512 bf16
  float*  gbuf = (float*)(smem + 128*HH*2);           // 128 x 8
  float*  cst  = gbuf + 128*8;                        // 32 x 8 cell state

  const int tid = threadIdx.x;
  const int dir = blockIdx.x >> 4;
  const int wg  = blockIdx.x & 15;
  const int j0  = wg * 32;

  const float*  GxD = Gx + (size_t)dir*MTOT*G4H;
  const __bf16* WD  = whh_pk + (size_t)dir*G4H*HH;
  __bf16*       hc  = hcur + (size_t)dir*16*HH;
  float*        hD  = hist + (size_t)dir*MTOT*HH;
  unsigned*     cn  = cnt + dir*1024;

  // Stage this WG's 128 Whh gate-rows into LDS (b128 chunks)
  for (int c = tid; c < 128*64; c += 256) {
    int n  = c >> 6;
    int k8 = (c & 63) << 3;
    int gr = (n >> 5)*HH + j0 + (n & 31);
    *(bf16x8*)(whhS + (size_t)n*HH + k8) = *(const bf16x8*)(WD + (size_t)gr*HH + k8);
  }
  cst[tid] = 0.f;                                     // 256 == 32*8
  __syncthreads();

  const int lane = tid & 31, wid = tid >> 5;
  const int lr = lane & 15, kb = (lane >> 4) << 3;
  const int nLoc = wid*16 + lr;                       // 0..127 gate-row in slice
  const __bf16* Bp  = whhS + (size_t)nLoc*HH;
  const __bf16* Apl = hc + (size_t)lr*HH;
  const int bb = tid & 7, uu = tid >> 3;              // pointwise (batch, unit)
  const v8f vzero = {0.f,0.f,0.f,0.f,0.f,0.f,0.f,0.f};

  for (int step = 0; step < TT; ++step) {
    const int t = dir ? (TT - 1 - step) : step;
    if (step > 0) {
      if (tid == 0) {
        while (__hip_atomic_load(&cn[step-1], __ATOMIC_ACQUIRE,
                                 __HIP_MEMORY_SCOPE_AGENT) < 16u)
          __builtin_amdgcn_s_sleep(1);
      }
      __syncthreads();
      __threadfence();                                // invalidate near caches
    }

    v8f acc = vzero;
#pragma unroll 4
    for (int k0 = 0; k0 < HH; k0 += 32) {
      bf16x8 alo = *(const bf16x8*)(Apl + k0 + kb);
      bf16x8 ahi = *(const bf16x8*)(Apl + k0 + 16 + kb);
      bf16x8 blo = *(const bf16x8*)(Bp  + k0 + kb);
      bf16x8 bhi = *(const bf16x8*)(Bp  + k0 + 16 + kb);
      v16bf av = CAT16(alo, ahi);
      v16bf bv = CAT16(blo, bhi);
      acc = __builtin_amdgcn_wmma_f32_16x16x32_bf16(
          false, av, false, bv, (short)0, acc, false, false);
    }
    if (lane < 16) {                                  // rows m=0..7 (valid batch)
#pragma unroll
      for (int i = 0; i < 8; ++i) gbuf[nLoc*8 + i] = acc[i];
    }
    __syncthreads();

    {
      size_t grow = (size_t)(bb*TT + t)*G4H + j0 + uu;
      float gi = gbuf[(0*32 + uu)*8 + bb] + GxD[grow + 0*HH];
      float gf = gbuf[(1*32 + uu)*8 + bb] + GxD[grow + 1*HH];
      float gg = gbuf[(2*32 + uu)*8 + bb] + GxD[grow + 2*HH];
      float go = gbuf[(3*32 + uu)*8 + bb] + GxD[grow + 3*HH];
      float si = 1.f/(1.f + __expf(-gi));
      float sf = 1.f/(1.f + __expf(-gf));
      float so = 1.f/(1.f + __expf(-go));
      float c  = sf*cst[uu*8 + bb] + si*tanhf(gg);
      float h  = so*tanhf(c);
      cst[uu*8 + bb] = c;
      hD[(size_t)(bb*TT + t)*HH + j0 + uu] = h;
      hc[(size_t)bb*HH + j0 + uu] = (__bf16)h;
    }
    __threadfence();                                  // publish h slice
    __syncthreads();
    if (tid == 0)
      __hip_atomic_fetch_add(&cn[step], 1u, __ATOMIC_RELEASE,
                             __HIP_MEMORY_SCOPE_AGENT);
  }
}

// ---------------------------------------------------------------------------
// Final FC + sigmoid: one wave per (b,t)
// ---------------------------------------------------------------------------
__global__ __launch_bounds__(256) void k_fc(
    const float* __restrict__ hist, const float* __restrict__ Wfc,
    const float* __restrict__ bfc, float* __restrict__ out)
{
  int wid = threadIdx.x >> 5, lane = threadIdx.x & 31;
  int m = blockIdx.x*8 + wid;
  if (m >= MTOT) return;
  const float* hf = hist + (size_t)m*HH;
  const float* hb = hist + (size_t)MTOT*HH + (size_t)m*HH;
  float s = 0.f;
  for (int h = lane; h < HH; h += 32)
    s += hf[h]*Wfc[h] + hb[h]*Wfc[HH + h];
  for (int o = 16; o > 0; o >>= 1) s += __shfl_xor(s, o, 32);
  if (lane == 0) out[m] = 1.f/(1.f + __expf(-(s + bfc[0])));
}

// ---------------------------------------------------------------------------
// Host
// ---------------------------------------------------------------------------
static inline char* carve(char*& p, size_t bytes) {
  char* r = p;
  p += (bytes + 255) & ~((size_t)255);
  return r;
}

extern "C" void kernel_launch(void* const* d_in, const int* in_sizes, int n_in,
                              void* d_out, int out_size, void* d_ws, size_t ws_size,
                              hipStream_t stream) {
  (void)in_sizes; (void)n_in; (void)out_size; (void)ws_size;
  const float* x      = (const float*)d_in[0];
  const float* Wdb    = (const float*)d_in[1];
  const float* bdb    = (const float*)d_in[2];
  const float* Wda    = (const float*)d_in[3];
  const float* bda    = (const float*)d_in[4];
  const float* Wr     = (const float*)d_in[5];
  const float* br     = (const float*)d_in[6];
  const float* Wih_fw = (const float*)d_in[7];
  const float* Whh_fw = (const float*)d_in[8];
  const float* bih_fw = (const float*)d_in[9];
  const float* bhh_fw = (const float*)d_in[10];
  const float* Wih_bw = (const float*)d_in[11];
  const float* Whh_bw = (const float*)d_in[12];
  const float* bih_bw = (const float*)d_in[13];
  const float* bhh_bw = (const float*)d_in[14];
  const float* Wfc    = (const float*)d_in[15];
  const float* bfc    = (const float*)d_in[16];
  float* out = (float*)d_out;

  // Workspace carving (~310 MB total)
  char* p = (char*)d_ws;
  __bf16* xh      = (__bf16*)carve(p, (size_t)BB*LL*DD*2);          // 33.6 MB
  __bf16* wpdb    = (__bf16*)carve(p, (size_t)HH*KDB*2);            // 10.5 MB
  __bf16* wpda    = (__bf16*)carve(p, (size_t)HH*KDB*2);            // 10.5 MB
  __bf16* wpr     = (__bf16*)carve(p, (size_t)HH*KR*2);             // 21.0 MB
  float*  db_o    = (float*) carve(p, (size_t)MTOT*HH*4);           // 16.6 MB
  float*  da_o    = (float*) carve(p, (size_t)MTOT*HH*4);           // 16.6 MB
  float*  r_o     = (float*) carve(p, (size_t)MTOT*HH*4);           // 16.6 MB
  __bf16* jointh  = (__bf16*)carve(p, (size_t)MTOT*KJP*2);          //  8.8 MB
  __bf16* wih_pk  = (__bf16*)carve(p, (size_t)2*G4H*KJP*2);         //  4.5 MB
  float*  bias_c  = (float*) carve(p, (size_t)2*G4H*4);             //  16 KB
  float*  Gx      = (float*) carve(p, (size_t)2*MTOT*G4H*4);        // 133  MB
  __bf16* whh_pk  = (__bf16*)carve(p, (size_t)2*G4H*HH*2);          //  4.2 MB
  float*  hist    = (float*) carve(p, (size_t)2*MTOT*HH*4);         // 33.2 MB
  __bf16* hcur    = (__bf16*)carve(p, (size_t)2*16*HH*2);           //  32 KB
  unsigned* cnt   = (unsigned*)carve(p, (size_t)2*1024*4);          //   8 KB

  // 0) per-call re-init (counters + padded h state)
  k_init<<<64, 256, 0, stream>>>(cnt, hcur);

  // 1) conversions / packing
  k_cvt_bf16 <<<4096, 256, 0, stream>>>(x, xh, BB*LL*DD);
  k_pack_convw<<<2048, 256, 0, stream>>>(Wdb, wpdb, OM);
  k_pack_convw<<<2048, 256, 0, stream>>>(Wda, wpda, OM);
  k_pack_convw<<<4096, 256, 0, stream>>>(Wr,  wpr,  2*OM);
  k_pack_wih <<<1024, 256, 0, stream>>>(Wih_fw, wih_pk);
  k_pack_wih <<<1024, 256, 0, stream>>>(Wih_bw, wih_pk + (size_t)G4H*KJP);
  k_cvt_bf16 <<<1024, 256, 0, stream>>>(Whh_fw, whh_pk, G4H*HH);
  k_cvt_bf16 <<<1024, 256, 0, stream>>>(Whh_bw, whh_pk + (size_t)G4H*HH, G4H*HH);
  k_bias2    <<<8, 256, 0, stream>>>(bih_fw, bhh_fw, bias_c, G4H);
  k_bias2    <<<8, 256, 0, stream>>>(bih_bw, bhh_bw, bias_c + G4H, G4H);

  // 2) conv GEMMs (windows of xh are contiguous in K)
  dim3 gc((MTOT + 63)/64, HH/256);
  k_gemm<<<gc, 256, 0, stream>>>(xh,          (long)LL*DD, (long)DD, TT, MTOT,
                                 wpdb, bdb, db_o, HH, KDB);
  k_gemm<<<gc, 256, 0, stream>>>(xh + OM*DD,  (long)LL*DD, (long)DD, TT, MTOT,
                                 wpda, bda, da_o, HH, KDB);
  k_gemm<<<gc, 256, 0, stream>>>(xh,          (long)LL*DD, (long)DD, TT, MTOT,
                                 wpr,  br,  r_o,  HH, KR);

  // 3) vts + joint assembly
  k_assemble<<<MTOT, 256, 0, stream>>>(db_o, da_o, r_o, jointh);

  // 4) input projections Gx = joint @ Wih^T + (bih+bhh)
  dim3 gj((MTOT + 63)/64, G4H/256);
  k_gemm<<<gj, 256, 0, stream>>>(jointh, 0L, (long)KJP, MTOT, MTOT,
                                 wih_pk, bias_c, Gx, G4H, KJP);
  k_gemm<<<gj, 256, 0, stream>>>(jointh, 0L, (long)KJP, MTOT, MTOT,
                                 wih_pk + (size_t)G4H*KJP, bias_c + G4H,
                                 Gx + (size_t)MTOT*G4H, G4H, KJP);

  // 5) persistent bidirectional recurrence (32 WGs, 136KB dynamic LDS each)
  const int smemBytes = 128*HH*2 + 128*8*4 + 32*8*4;   // 136,192 B
  (void)hipFuncSetAttribute((const void*)k_lstm,
                            hipFuncAttributeMaxDynamicSharedMemorySize, smemBytes);
  k_lstm<<<32, 256, smemBytes, stream>>>(Gx, whh_pk, hcur, hist, cnt);

  // 6) final FC + sigmoid
  k_fc<<<(MTOT + 7)/8, 256, 0, stream>>>(hist, Wfc, bfc, out);
}
